// ObjectAwareTokenizer_87239375716507
// MI455X (gfx1250) — matrix operand order repaired and checked by reading, not compiled, not run on hardware
//
#include <hip/hip_runtime.h>

// ---------------------------------------------------------------------------
// ObjectAwareTokenizer on gfx1250 (CDNA5, wave32, WMMA f16 16x16x32)
//
// Folded formulation:
//   t   = relu(conv3x3(x) + b1)                  (im2col GEMM, K=99 pad 128)
//   out = Weff @ t + beff + occ * contrib[h%8,w%8]
// where Weff = (I + fus_w[:, :64]) @ conv2_w,  beff = (I+F_pe)@b2 + fus_b,
// and contrib folds the (constant) shape/rel-pos patterns through fus_w.
// ---------------------------------------------------------------------------

typedef __attribute__((ext_vector_type(16))) _Float16 v16h;
typedef __attribute__((ext_vector_type(8)))  _Float16 v8h;
typedef __attribute__((ext_vector_type(8)))  float    v8f;

#define NB   32
#define NH   128
#define NW   128
#define CIN  11
#define ED   64
#define SD   32

// padded LDS strides (halfs), all multiples of 8 halfs (16B) for b128 loads,
// chosen to rotate banks between rows.
#define IMS  136   // im2col activation rows: 128 k (99 valid) + pad
#define W1S  136   // conv1 weights o-major
#define WES  80    // Weff o-major (64 k + pad)
#define TSS  72    // t rows (64 k + pad)

__device__ __forceinline__ v16h ld_pair(const _Float16* p0, const _Float16* p1) {
    v8h lo = *(const v8h*)p0;
    v8h hi = *(const v8h*)p1;
    return __builtin_shufflevector(lo, hi, 0,1,2,3,4,5,6,7,8,9,10,11,12,13,14,15);
}

// ---------------------------------------------------------------------------
// Precompute (1 block): shape encoder, rel-pos, contrib, Weff/beff, f16 packs
// ---------------------------------------------------------------------------
__global__ __launch_bounds__(256) void oat_precompute(
    const float* __restrict__ conv1_w, const float* __restrict__ conv2_w,
    const float* __restrict__ conv2_b, const float* __restrict__ se_conv_w,
    const float* __restrict__ se_conv_b, const float* __restrict__ se_lin_w,
    const float* __restrict__ se_lin_b, const float* __restrict__ rp_w,
    const float* __restrict__ rp_b, const float* __restrict__ fus_w,
    const float* __restrict__ fus_b,
    float* __restrict__ contrib,   // [64 pp][64 o]
    float* __restrict__ beff,      // [64]
    _Float16* __restrict__ W1A,    // [64 o][128 k]
    _Float16* __restrict__ WeffA)  // [64 o][64 k]
{
    __shared__ float sh_h[16 * 49];   // se conv output, relu'd
    __shared__ float sh_pool[256];    // adaptive maxpool 7->4, flattened C-major
    __shared__ float sh_svec[SD];     // shape_vec

    const int tid = threadIdx.x;

    // se conv3x3 (pad 1) on all-ones 7x7 mask, relu
    for (int i = tid; i < 16 * 49; i += 256) {
        int c = i / 49, r = i % 49, py = r / 7, px = r % 7;
        float s = se_conv_b[c];
        #pragma unroll
        for (int di = 0; di < 3; ++di)
            #pragma unroll
            for (int dj = 0; dj < 3; ++dj) {
                int iy = py - 1 + di, ix = px - 1 + dj;
                if (iy >= 0 && iy < 7 && ix >= 0 && ix < 7)
                    s += se_conv_w[c * 9 + di * 3 + dj];
            }
        sh_h[i] = s > 0.f ? s : 0.f;
    }
    __syncthreads();

    // adaptive maxpool 7->4 (torch bin edges), C-major flatten
    if (tid < 256) {
        const int bs[4] = {0, 1, 3, 5}, be[4] = {2, 4, 6, 7};
        int c = tid / 16, r4 = (tid / 4) % 4, c4 = tid % 4;
        float m = -3.4e38f;
        for (int rs = bs[r4]; rs < be[r4]; ++rs)
            for (int cs = bs[c4]; cs < be[c4]; ++cs) {
                float v = sh_h[c * 49 + rs * 7 + cs];
                m = v > m ? v : m;
            }
        sh_pool[tid] = m;
    }
    __syncthreads();

    // shape_vec = relu(linear(pooled))
    if (tid < SD) {
        float s = se_lin_b[tid];
        for (int k = 0; k < 256; ++k) s += se_lin_w[tid * 256 + k] * sh_pool[k];
        sh_svec[tid] = s > 0.f ? s : 0.f;
    }
    __syncthreads();

    // contrib[pp][o] = F_s @ pat_s + F_p @ pat_p  (zero outside 7x7 region)
    for (int i = tid; i < 64 * 64; i += 256) {
        int pp = i >> 6, o = i & 63;
        int py = pp >> 3, px = pp & 7;
        float acc = 0.f;
        if (py < 7 && px < 7) {
            float yy = (float)py * (1.f / 6.f), xx = (float)px * (1.f / 6.f);
            for (int s = 0; s < SD; ++s) {
                float pos = rp_w[2 * s] * yy + rp_w[2 * s + 1] * xx + rp_b[s];
                acc += fus_w[o * 128 + 64 + s] * sh_svec[s]
                     + fus_w[o * 128 + 96 + s] * pos;
            }
        }
        contrib[i] = acc;
    }

    // Weff[o][k] = sum_c (fus_w[o,c] + (c==o)) * conv2_w[c][k]   (f16 pack)
    for (int i = tid; i < 64 * 64; i += 256) {
        int o = i >> 6, k = i & 63;
        float acc = 0.f;
        for (int c = 0; c < ED; ++c) {
            float m = fus_w[o * 128 + c] + (c == o ? 1.f : 0.f);
            acc += m * conv2_w[c * ED + k];
        }
        WeffA[i] = (_Float16)acc;
    }

    // beff[o] = fus_b[o] + sum_c (fus_w[o,c] + (c==o)) * conv2_b[c]
    if (tid < ED) {
        float acc = fus_b[tid];
        for (int c = 0; c < ED; ++c)
            acc += (fus_w[tid * 128 + c] + (c == tid ? 1.f : 0.f)) * conv2_b[c];
        beff[tid] = acc;
    }

    // conv1 weights, o-major, K padded 99 -> 128, f16
    for (int i = tid; i < 64 * 128; i += 256) {
        int o = i >> 7, k = i & 127;
        W1A[i] = (k < 99) ? (_Float16)conv1_w[o * 99 + k] : (_Float16)0.f;
    }
}

// ---------------------------------------------------------------------------
// Fused main kernel: one workgroup = one (b, h) row of 128 pixels, 8 waves.
// GEMM1: t[64ch x 128pix] = relu(W1[64x128k] * im2col[128k x 128pix] + b1)
// GEMM2: out = Weff[64x64] * t + beff + occ * contrib  (coalesced stores)
// ---------------------------------------------------------------------------
__global__ __launch_bounds__(256, 2) void oat_fused(
    const float* __restrict__ x, const float* __restrict__ conv1_b,
    const float* __restrict__ contrib, const float* __restrict__ beff,
    const _Float16* __restrict__ W1A, const _Float16* __restrict__ WeffA,
    float* __restrict__ out)
{
    __shared__ __align__(16) _Float16 im[NW * IMS];   // im2col rows  [pix][k]
    __shared__ __align__(16) _Float16 w1s[ED * W1S];  // conv1 wts    [o][k]
    __shared__ __align__(16) _Float16 wes[ED * WES];  // Weff         [o][k]
    __shared__ __align__(16) _Float16 ts[NW * TSS];   // t            [pix][k]
    __shared__ float cts[8 * 64];                     // contrib row  [w%8][o]
    __shared__ float occs[NW];
    __shared__ float beffs[ED];
    __shared__ float b1s[ED];

    const int h = blockIdx.x, b = blockIdx.y;
    const int tid = threadIdx.x;
    const int lane = tid & 31, wave = tid >> 5;
    const int ln15 = lane & 15, hi = lane >> 4;   // wave32 half-select

    // ---- phase 0: stage weights / constants / im2col into LDS -------------
    for (int i = tid; i < ED * 128; i += 256) {
        int o = i >> 7, k = i & 127;
        w1s[o * W1S + k] = W1A[i];
    }
    for (int i = tid; i < ED * ED; i += 256) {
        int o = i >> 6, k = i & 63;
        wes[o * WES + k] = WeffA[i];
    }
    for (int i = tid; i < 512; i += 256) cts[i] = contrib[(h & 7) * 512 + i];
    if (tid < ED) { beffs[tid] = beff[tid]; b1s[tid] = conv1_b[tid]; }
    if (tid < NW) {
        // one-hot input: class 0 <=> channel 0 == 1  =>  occ = !channel0
        float x0 = x[(((size_t)b * CIN) * NH + h) * NW + tid];
        occs[tid] = (x0 > 0.5f) ? 0.f : 1.f;
    }
    // im2col: k = (c*3 + di)*3 + dj, 99 valid, zero-padded to 128
    for (int i = tid; i < 128 * NW; i += 256) {
        int k = i >> 7, w = i & 127;
        float v = 0.f;
        if (k < 99) {
            int c = k / 9, r = k % 9, di = r / 3, dj = r % 3;
            int hh = h - 1 + di, ww = w - 1 + dj;
            if (hh >= 0 && hh < NH && ww >= 0 && ww < NW)
                v = x[(((size_t)b * CIN + c) * NH + hh) * NW + ww];
        }
        im[w * IMS + k] = (_Float16)v;
    }
    __syncthreads();

    const int pixBase = wave * 16;      // each wave owns 16 pixels (N tiles)
    const int pix = pixBase + ln15;

    // ---- GEMM1: 4 mTiles (64 out ch) x this wave's pixel tile, K=128 ------
    {
        v8f acc[4] = {};
        #pragma unroll
        for (int kc = 0; kc < 4; ++kc) {
            const int kb = kc * 32;
            // B fragment (activation, pixel-major): 16 contiguous halfs
            const _Float16* bp = &im[pix * IMS + kb + hi * 16];
            v16h bfrag = ld_pair(bp, bp + 8);
            #pragma unroll
            for (int mT = 0; mT < 4; ++mT) {
                // A fragment (weights, o-major): 16 contiguous halfs
                const _Float16* ap = &w1s[(mT * 16 + ln15) * W1S + kb + hi * 16];
                v16h afrag = ld_pair(ap, ap + 8);
                acc[mT] = __builtin_amdgcn_wmma_f32_16x16x32_f16(
                    false, afrag, false, bfrag, (short)0, acc[mT], false, false);
            }
        }
        // bias + relu, pack to f16 t rows (this wave's 16 pixels only)
        #pragma unroll
        for (int mT = 0; mT < 4; ++mT)
            #pragma unroll
            for (int v = 0; v < 8; ++v) {
                int ch = mT * 16 + v + hi * 8;
                float val = acc[mT][v] + b1s[ch];
                ts[pix * TSS + ch] = (_Float16)(val > 0.f ? val : 0.f);
            }
    }
    __syncthreads();

    // ---- GEMM2: out = Weff @ t + beff + occ*contrib, K=64 -----------------
    {
        v8f acc[4] = {};
        #pragma unroll
        for (int kc = 0; kc < 2; ++kc) {
            const int kb = kc * 32;
            const _Float16* bp = &ts[pix * TSS + kb + hi * 16];
            v16h bfrag = ld_pair(bp, bp + 8);
            #pragma unroll
            for (int mT = 0; mT < 4; ++mT) {
                const _Float16* ap = &wes[(mT * 16 + ln15) * WES + kb + hi * 16];
                v16h afrag = ld_pair(ap, ap + 8);
                acc[mT] = __builtin_amdgcn_wmma_f32_16x16x32_f16(
                    false, afrag, false, bfrag, (short)0, acc[mT], false, false);
            }
        }
        const float occ = occs[pix];
        const float* ctrow = &cts[(pix & 7) * 64];
        #pragma unroll
        for (int mT = 0; mT < 4; ++mT)
            #pragma unroll
            for (int v = 0; v < 8; ++v) {
                int o = mT * 16 + v + hi * 8;
                float val = acc[mT][v] + beffs[o] + occ * ctrow[o];
                // lanes 0-15 cover 16 consecutive w  => coalesced b32 stores
                out[((((size_t)b * ED + o) * NH + h) * NW) + pix] = val;
            }
    }
}

// ---------------------------------------------------------------------------
extern "C" void kernel_launch(void* const* d_in, const int* in_sizes, int n_in,
                              void* d_out, int out_size, void* d_ws, size_t ws_size,
                              hipStream_t stream) {
    const float* x        = (const float*)d_in[0];
    const float* conv1_w  = (const float*)d_in[1];
    const float* conv1_b  = (const float*)d_in[2];
    const float* conv2_w  = (const float*)d_in[3];
    const float* conv2_b  = (const float*)d_in[4];
    const float* se_conv_w= (const float*)d_in[5];
    const float* se_conv_b= (const float*)d_in[6];
    const float* se_lin_w = (const float*)d_in[7];
    const float* se_lin_b = (const float*)d_in[8];
    const float* rp_w     = (const float*)d_in[9];
    const float* rp_b     = (const float*)d_in[10];
    const float* fus_w    = (const float*)d_in[11];
    const float* fus_b    = (const float*)d_in[12];

    // workspace layout
    float* ws_f      = (float*)d_ws;
    float* contrib   = ws_f;                       // 4096 f32
    float* beff      = ws_f + 4096;                // 64   f32
    _Float16* W1A    = (_Float16*)(ws_f + 4160);   // 8192 f16
    _Float16* WeffA  = W1A + 64 * 128;             // 4096 f16

    oat_precompute<<<1, 256, 0, stream>>>(
        conv1_w, conv2_w, conv2_b, se_conv_w, se_conv_b, se_lin_w, se_lin_b,
        rp_w, rp_b, fus_w, fus_b, contrib, beff, W1A, WeffA);

    dim3 grid(NH, NB);   // one workgroup per (h, b) pixel row
    oat_fused<<<grid, 256, 0, stream>>>(
        x, conv1_b, contrib, beff, W1A, WeffA, (float*)d_out);
}
